// Model_63556926046558
// MI455X (gfx1250) — matrix operand order repaired and checked by reading, not compiled, not run on hardware
//
#include <hip/hip_runtime.h>
#include <math.h>

// ---------------------------------------------------------------------------
// Sliding-window causal attention (W=512) with qk-norm + RoPE, for MI455X.
// Kernel 1: normalize+RoPE Q/K -> f16, V -> f16 transposed [b,h,a,s].
// Kernel 2: streaming attention, 32 keys/step, v_wmma_f32_16x16x32_f16.
//           qk-norm bounds |scores| <= 0.125  =>  softmax with fixed max=0,
//           no online-max bookkeeping needed. Masking only on edge chunks.
// ---------------------------------------------------------------------------

constexpr int Hn = 16;          // heads
constexpr int Ad = 64;          // head dim
constexpr int Wl = 512;         // window length
constexpr int Sq = 4096;        // sequence length
constexpr int Bb = 2;           // batch
constexpr int Dm = Hn * Ad;     // 1024

typedef _Float16 half8  __attribute__((ext_vector_type(8)));
typedef _Float16 half16 __attribute__((ext_vector_type(16)));
typedef float    float8 __attribute__((ext_vector_type(8)));

static __device__ __forceinline__ float8 wmma16x16x32(half16 a, half16 b, float8 c) {
  // D = A(16x32 f16) * B(32x16 f16) + C(16x16 f32)
  return __builtin_amdgcn_wmma_f32_16x16x32_f16(
      /*neg_a=*/false, a, /*neg_b=*/false, b,
      /*c_mod=*/(short)0, c, /*reuse_a=*/false, /*reuse_b=*/false);
}

static __device__ __forceinline__ half16 cat8(half8 lo, half8 hi) {
  return __builtin_shufflevector(lo, hi, 0, 1, 2, 3, 4, 5, 6, 7,
                                 8, 9, 10, 11, 12, 13, 14, 15);
}

// ---------------------------------------------------------------------------
// Kernel 1: per (b,s,h) row of 64: L2-normalize + RoPE for Q,K (f16 out),
// and transpose-store V as f16 [bh][a][s].
// ---------------------------------------------------------------------------
__global__ __launch_bounds__(256) void prep_rope_norm(
    const float* __restrict__ xq, const float* __restrict__ xk,
    const float* __restrict__ xv,
    _Float16* __restrict__ qh, _Float16* __restrict__ kh,
    _Float16* __restrict__ vt)
{
  const int rid = blockIdx.x * 256 + threadIdx.x;   // ((b*Sq + s)*Hn + h)
  if (rid >= Bb * Sq * Hn) return;
  const int h  = rid & (Hn - 1);
  const int bs = rid >> 4;
  const int s  = bs & (Sq - 1);
  const int b  = bs >> 12;
  const int bh = b * Hn + h;
  const size_t src = (size_t)rid * Ad;   // input row is contiguous in [b,s,d]

  // RoPE tables: inv_freq[i] = 10000^(-(2i)/64); angle = s * inv_freq[i]
  float cs[Ad / 2], sn[Ad / 2];
#pragma unroll
  for (int i = 0; i < Ad / 2; ++i) {
    const float fr = (float)s * __expf(-(float)(2 * i) * (9.2103403720f / (float)Ad));
    sincosf(fr, &sn[i], &cs[i]);
  }

  const float* srcs[2] = { xq + src, xk + src };
  _Float16* dsts[2] = { qh + ((size_t)bh * Sq + s) * Ad,
                        kh + ((size_t)bh * Sq + s) * Ad };
#pragma unroll
  for (int t = 0; t < 2; ++t) {
    float x[Ad];
    float ssum = 0.f;
#pragma unroll
    for (int a = 0; a < Ad; ++a) { x[a] = srcs[t][a]; ssum += x[a] * x[a]; }
    const float inv = 1.f / fmaxf(sqrtf(ssum), 1e-6f);
#pragma unroll
    for (int i = 0; i < Ad / 2; ++i) {
      const float x1 = x[i] * inv, x2 = x[i + Ad / 2] * inv;
      dsts[t][i]          = (_Float16)( x1 * cs[i] + x2 * sn[i]);
      dsts[t][i + Ad / 2] = (_Float16)(-x1 * sn[i] + x2 * cs[i]);
    }
  }
  // V transposed: vt[bh][a][s]
#pragma unroll
  for (int a = 0; a < Ad; ++a)
    vt[((size_t)bh * Ad + a) * Sq + s] = (_Float16)xv[src + a];
}

// ---------------------------------------------------------------------------
// One 32-key step. MASKED variant applies the causal/window mask and clamps
// addresses; the unmasked variant is pure loads + 8 WMMAs + exp.
// Softmax uses fixed max = 0 (scores bounded by 0.125 thanks to qk-norm).
// ---------------------------------------------------------------------------
template <bool MASKED>
static __device__ __forceinline__ void attn_step(
    int kb, int qg, int nq, int khf,
    const _Float16* __restrict__ K, const _Float16* __restrict__ V,
    half16 bq0, half16 bq1, float8 (&acc)[4], float& l_run)
{
  constexpr float scale = 0.125f;   // 1/sqrt(64)

  // ---- K fragment addresses (A-layout rows = keys) ----
  int kr0 = kb + nq;
  int kr1 = kb + 16 + nq;
  if (MASKED) {
    if (kr0 > Sq - 1) kr0 = Sq - 1;
    if (kr1 > Sq - 1) kr1 = Sq - 1;
  }
  const _Float16* kp0 = K + (size_t)kr0 * Ad + khf * 8;
  const _Float16* kp1 = K + (size_t)kr1 * Ad + khf * 8;

  // ---- batch all K loads, then the 4 score WMMAs ----
  const half16 ak00 = cat8(*(const half8*)(kp0),      *(const half8*)(kp0 + 16));
  const half16 ak01 = cat8(*(const half8*)(kp0 + 32), *(const half8*)(kp0 + 48));
  const half16 ak10 = cat8(*(const half8*)(kp1),      *(const half8*)(kp1 + 16));
  const half16 ak11 = cat8(*(const half8*)(kp1 + 32), *(const half8*)(kp1 + 48));

  float8 s0 = {}, s1 = {};
  s0 = wmma16x16x32(ak00, bq0, s0);
  s0 = wmma16x16x32(ak01, bq1, s0);
  s1 = wmma16x16x32(ak10, bq0, s1);
  s1 = wmma16x16x32(ak11, bq1, s1);

  // ---- issue V loads now; they overlap the softmax VALU work ----
  int kbv = kb;
  if (MASKED) { if (kbv > Sq - 32) kbv = Sq - 32; }
  half16 av[4];
#pragma unroll
  for (int t = 0; t < 4; ++t) {
    const _Float16* vp = V + (size_t)(t * 16 + nq) * Sq + kbv + khf * 16;
    av[t] = *(const half16*)(vp);
  }

  // ---- p = exp(score) (fixed max = 0); mask only on edge chunks ----
  float p0[8], p1[8];
  float psum = 0.f;
#pragma unroll
  for (int r = 0; r < 8; ++r) {
    const float x0 = s0[r] * scale;
    const float x1 = s1[r] * scale;
    if (MASKED) {
      const int kv0 = kb + r + khf * 8;
      const int kv1 = kv0 + 16;
      const bool v0 = (kv0 <= qg) && (qg - kv0 < Wl);
      const bool v1 = (kv1 <= qg) && (qg - kv1 < Wl);
      p0[r] = v0 ? __expf(x0) : 0.f;
      p1[r] = v1 ? __expf(x1) : 0.f;
    } else {
      p0[r] = __expf(x0);
      p1[r] = __expf(x1);
    }
    psum += p0[r] + p1[r];
  }
  psum += __shfl_xor(psum, 16);
  l_run += psum;

  // ---- assemble Pᵀ as B fragment (32 keys x 16 queries) ----
  // A lane needs only its partner's p0 (khf=0) or p1 (khf=1): pre-select the
  // value the partner wants and do ONE shuffle per r instead of two.
  half16 bp;
#pragma unroll
  for (int r = 0; r < 8; ++r) {
    const float send = khf ? p0[r] : p1[r];       // what my partner needs
    const float recv = __shfl_xor(send, 16);      // what I need from partner
    bp[r]     = (_Float16)(khf ? recv  : p0[r]);  // keys kb+khf*16 + r
    bp[r + 8] = (_Float16)(khf ? p1[r] : recv);   // keys kb+khf*16 + 8 + r
  }

  // ---- PV: outᵀ[a,q] += Vᵀ[a,k] · Pᵀ[k,q] ----
#pragma unroll
  for (int t = 0; t < 4; ++t)
    acc[t] = wmma16x16x32(av[t], bp, acc[t]);
}

// ---------------------------------------------------------------------------
// Kernel 2: 128 threads = 4 waves per workgroup; each wave owns 16 queries.
// Sᵀ = K·Qᵀ (A = K rows, B = Q cols) so per-query stats are per-lane-column.
// ---------------------------------------------------------------------------
__global__ __launch_bounds__(128) void swa_attn_wmma(
    const _Float16* __restrict__ qh, const _Float16* __restrict__ kh,
    const _Float16* __restrict__ vt, float* __restrict__ out)
{
  const int lane = threadIdx.x & 31;
  const int wave = threadIdx.x >> 5;
  const int nq   = lane & 15;    // query column within tile / A-row index
  const int khf  = lane >> 4;    // lane-half selector

  const int wg  = blockIdx.x;          // Bb*Hn*64 workgroups
  const int bh  = wg >> 6;
  const int rem = wg & 63;             // qblock(3b) | sub(3b)
  const int qg0 = (rem >> 3) * Wl + (rem & 7) * 64 + wave * 16;

  const _Float16* Q = qh + (size_t)bh * Sq * Ad;
  const _Float16* K = kh + (size_t)bh * Sq * Ad;
  const _Float16* V = vt + (size_t)bh * Ad * Sq;

  // Q tile as B fragments (32x16): lane holds column nq; K idx = khf*8 + e (e<8), 16 + khf*8 + (e-8) (e>=8).
  const _Float16* qp = Q + (size_t)(qg0 + nq) * Ad + khf * 8;
  const half16 bq0 = cat8(*(const half8*)(qp),      *(const half8*)(qp + 16));   // a = 0..31
  const half16 bq1 = cat8(*(const half8*)(qp + 32), *(const half8*)(qp + 48));   // a = 32..63

  float8 acc[4] = {};                 // outᵀ tiles: a = t*16 + r + khf*8
  float l_run = 0.f;
  const int qg = qg0 + nq;

  int kstart = qg0 - (Wl - 1);
  if (kstart < 0) kstart = 0;
  kstart &= ~31;
  const int kend = qg0 + 15;

  // First fully-in-window chunk (aligned up) and last fully-causal chunk.
  int full_lo = (qg0 + 16 - Wl + 31) & ~31;
  if (full_lo < kstart) full_lo = kstart;
  const int full_hi = qg0 - 31;        // kb <= full_hi => all 32 keys <= all queries

  int kb = kstart;
  for (; kb < full_lo && kb <= kend; kb += 32)       // masked head (<= 2 chunks)
    attn_step<true>(kb, qg, nq, khf, K, V, bq0, bq1, acc, l_run);
  for (; kb <= full_hi; kb += 32)                    // unmasked interior (~14)
    attn_step<false>(kb, qg, nq, khf, K, V, bq0, bq1, acc, l_run);
  for (; kb <= kend; kb += 32)                       // masked tail (<= 2 chunks)
    attn_step<true>(kb, qg, nq, khf, K, V, bq0, bq1, acc, l_run);

  // ---- normalize and write out[b, s, h*64 + a] as 32-byte vector stores ----
  const float invl = 1.f / l_run;
  const int b = bh >> 4, h = bh & 15;
  float* op = out + ((size_t)b * Sq + qg) * Dm + h * Ad + khf * 8;
#pragma unroll
  for (int t = 0; t < 4; ++t) {
    float8 res;
#pragma unroll
    for (int r = 0; r < 8; ++r) res[r] = acc[t][r] * invl;
    *(float8*)(op + t * 16) = res;   // a = t*16 + khf*8 + r, contiguous in r
  }
}

// ---------------------------------------------------------------------------
extern "C" void kernel_launch(void* const* d_in, const int* in_sizes, int n_in,
                              void* d_out, int out_size, void* d_ws, size_t ws_size,
                              hipStream_t stream) {
  (void)in_sizes; (void)n_in; (void)out_size; (void)ws_size;
  const float* xq = (const float*)d_in[0];
  const float* xk = (const float*)d_in[1];
  const float* xv = (const float*)d_in[2];
  float* out = (float*)d_out;

  const size_t elems = (size_t)Bb * Hn * Sq * Ad;   // 8,388,608 halves each
  _Float16* qh = (_Float16*)d_ws;
  _Float16* kh = qh + elems;
  _Float16* vt = kh + elems;                        // ~50 MB total workspace

  dim3 g1((Bb * Sq * Hn + 255) / 256), b1(256);
  prep_rope_norm<<<g1, b1, 0, stream>>>(xq, xk, xv, qh, kh, vt);

  dim3 g2(Bb * Hn * 64), b2(128);                   // 4 waves / WG, 16 q / wave
  swa_attn_wmma<<<g2, b2, 0, stream>>>(qh, kh, vt, out);
}
